// LayerNorm_48773648614249
// MI455X (gfx1250) — matrix operand order repaired and checked by reading, not compiled, not run on hardware
//
#include <hip/hip_runtime.h>

constexpr int kD    = 768;
constexpr int kOut  = 196;
constexpr int kRows = 64;          // rows per block = 4 M-tiles
constexpr int kLdsStride = 776;    // 768 + 8 halves pad -> conflict-free
constexpr float kEps = 1e-6f;

typedef _Float16 v8h  __attribute__((ext_vector_type(8)));
typedef _Float16 v16h __attribute__((ext_vector_type(16)));
typedef float    v8f  __attribute__((ext_vector_type(8)));

__global__ void w_to_f16_kernel(const float* __restrict__ W,
                                _Float16* __restrict__ Wh, int n) {
    int i = blockIdx.x * blockDim.x + threadIdx.x;
    if (i < n) Wh[i] = (_Float16)W[i];
}

__global__ __launch_bounds__(256)
void ln_gemm_kernel(const float* __restrict__ x,
                    const float* __restrict__ gamma,
                    const float* __restrict__ beta,
                    const _Float16* __restrict__ Wh,
                    const float* __restrict__ bias,
                    float* __restrict__ out) {
    __shared__ _Float16 xn[kRows][kLdsStride];

    const int tid  = threadIdx.x;
    const int lane = tid & 31;
    const int wave = tid >> 5;
    const long row_base = (long)blockIdx.x * kRows;

    // ---------- LayerNorm: each wave normalizes 8 rows ----------
    for (int j = 0; j < 8; ++j) {
        const int r = wave * 8 + j;
        const float* xr = x + (row_base + r) * kD;
        float s = 0.f, ss = 0.f;
        #pragma unroll
        for (int i = 0; i < kD / 32; ++i) {
            float v = xr[lane + 32 * i];
            s  += v;
            ss += v * v;
        }
        #pragma unroll
        for (int off = 16; off >= 1; off >>= 1) {
            s  += __shfl_xor(s,  off, 32);
            ss += __shfl_xor(ss, off, 32);
        }
        const float mean = s * (1.0f / kD);
        const float var  = ss * (1.0f / kD) - mean * mean;
        const float rstd = rsqrtf(var + kEps);
        #pragma unroll
        for (int i = 0; i < kD / 32; ++i) {
            const int d = lane + 32 * i;
            const float v = (xr[d] - mean) * rstd * gamma[d] + beta[d];
            xn[r][d] = (_Float16)v;
        }
    }
    __syncthreads();

    // ---------- GEMM: each wave owns N-tiles wave and wave+8 (if < 13), all 4 M-tiles ----------
    // B fragment loaded once per K-step feeds 4 WMMAs (4x B-reuse).
    const int mrow = lane & 15;   // M (A/D) or N (B/D) sub-index
    const int hi   = lane >> 4;   // lane half select

    for (int nt = wave; nt < 13; nt += 8) {
        const int n_base = nt * 16;
        const int o = min(n_base + mrow, kOut - 1);   // clamp: no OOB, EXEC full

        const float bv = bias[o];
        v8f c[4];
        #pragma unroll
        for (int m = 0; m < 4; ++m)
            #pragma unroll
            for (int v = 0; v < 8; ++v) c[m][v] = bv;

        const _Float16* wrow = Wh + (long)o * kD;

        #pragma unroll 2
        for (int k0 = 0; k0 < kD; k0 += 32) {
            // B fragment (32x16): lane n holds 16 contiguous K from W row o
            const v16h bfr = *(const v16h*)(wrow + k0 + hi * 16);
            #pragma unroll
            for (int m = 0; m < 4; ++m) {
                // A fragment (contiguous k order, same as B):
                //   lanes 0-15: K = k0+0..15 of row M
                //   lanes16-31: K = k0+16..31 of row M
                const _Float16* arow = &xn[m * 16 + mrow][0];
                const v8h a0 = *(const v8h*)(arow + k0 + hi * 16);
                const v8h a1 = *(const v8h*)(arow + k0 + hi * 16 + 8);
                const v16h a = __builtin_shufflevector(a0, a1,
                    0,1,2,3,4,5,6,7,8,9,10,11,12,13,14,15);
                c[m] = __builtin_amdgcn_wmma_f32_16x16x32_f16(
                           false, a, false, bfr, (short)0, c[m], false, false);
            }
        }

        // D layout: lane l, vgpr v -> [M = v + 8*hi][N = l&15]
        const int oo = n_base + mrow;
        if (oo < kOut) {
            #pragma unroll
            for (int m = 0; m < 4; ++m) {
                float* orow = out + (row_base + m * 16 + hi * 8) * (long)kOut + oo;
                #pragma unroll
                for (int v = 0; v < 8; ++v) orow[(long)v * kOut] = c[m][v];
            }
        }
    }
}

extern "C" void kernel_launch(void* const* d_in, const int* in_sizes, int n_in,
                              void* d_out, int out_size, void* d_ws, size_t ws_size,
                              hipStream_t stream) {
    const float* x     = (const float*)d_in[0];
    const float* gamma = (const float*)d_in[1];
    const float* beta  = (const float*)d_in[2];
    const float* W     = (const float*)d_in[3];
    const float* b     = (const float*)d_in[4];
    float* out = (float*)d_out;

    _Float16* Wh = (_Float16*)d_ws;   // 196*768*2 = 301,056 bytes of scratch

    const int nW = kOut * kD;
    w_to_f16_kernel<<<(nW + 255) / 256, 256, 0, stream>>>(W, Wh, nW);

    const long total_rows = 16L * 4096;
    const int blocks = (int)(total_rows / kRows);   // 1024
    ln_gemm_kernel<<<blocks, 256, 0, stream>>>(x, gamma, beta, Wh, b, out);
}
